// ClassicalSelfAttention_65481071402315
// MI455X (gfx1250) — matrix operand order repaired and checked by reading, not compiled, not run on hardware
//
#include <hip/hip_runtime.h>
#include <hip/hip_bf16.h>
#include <stdint.h>

// Problem constants (from reference): B=4, N=4096, D=1024, H=16, HD=64
#define BB        4
#define NN        4096
#define DD        1024
#define HH        16
#define HDD       64
#define M_TOK     (BB * NN)     // 16384 tokens
#define THREE_D   (3 * DD)      // 3072

typedef __attribute__((ext_vector_type(16))) __bf16 v16bf;
typedef __attribute__((ext_vector_type(8)))  __bf16 v8bf;
typedef __attribute__((ext_vector_type(8)))  float  v8f;

static __device__ __forceinline__ unsigned short f32_to_bf16(float f) {
    unsigned int u = __builtin_bit_cast(unsigned int, f);
    u += 0x7FFFu + ((u >> 16) & 1u);          // round-to-nearest-even
    return (unsigned short)(u >> 16);
}

// ---------------------------------------------------------------------------
// WMMA fragment loaders (CDNA5 16x16x32 bf16 layouts, ISA 7.12.2)
// A (16x32, row-major along K): lane (r=lane&15, hi=lane>>4) holds
//   elems 0..7  = A[r, k0 + hi*8 .. +8)
//   elems 8..15 = A[r, k0 + 16 + hi*8 .. +8)
// B (32x16, column n stored as row n of W[N,K]): lane (c, hi) holds
//   elems 0..7  = W[c, k0 + hi*8 .. +8)
//   elems 8..15 = W[c, k0 + 16 + hi*8 .. +8)
// ---------------------------------------------------------------------------
static __device__ __forceinline__ v16bf load_a_frag(const unsigned short* base,
                                                    int ld, int k0, int lane) {
    const int r  = lane & 15;
    const int hi = lane >> 4;
    const unsigned short* p = base + (size_t)r * ld + k0 + hi * 8;
    v8bf c0 = *(const v8bf*)p;
    v8bf c1 = *(const v8bf*)(p + 16);
    v16bf out;
#pragma unroll
    for (int i = 0; i < 8; ++i) { out[i] = c0[i]; out[i + 8] = c1[i]; }
    return out;
}

static __device__ __forceinline__ v16bf load_b_frag(const unsigned short* base,
                                                    int ld, int k0, int lane) {
    const int c  = lane & 15;
    const int hi = lane >> 4;
    const unsigned short* p = base + (size_t)c * ld + k0 + hi * 8;
    v8bf c0 = *(const v8bf*)p;
    v8bf c1 = *(const v8bf*)(p + 16);
    v16bf out;
#pragma unroll
    for (int i = 0; i < 8; ++i) { out[i] = c0[i]; out[i + 8] = c1[i]; }
    return out;
}

// ---------------------------------------------------------------------------
// fp32 -> bf16 conversion (vectorized 4-wide)
// ---------------------------------------------------------------------------
__global__ void f32_to_bf16_kernel(const float* __restrict__ in,
                                   unsigned short* __restrict__ out, int n4) {
    int i = blockIdx.x * blockDim.x + threadIdx.x;
    if (i < n4) {
        float4 f = ((const float4*)in)[i];
        ushort4 o;
        o.x = f32_to_bf16(f.x); o.y = f32_to_bf16(f.y);
        o.z = f32_to_bf16(f.z); o.w = f32_to_bf16(f.w);
        ((ushort4*)out)[i] = o;
    }
}

// ---------------------------------------------------------------------------
// GEMM: C[M,Nc] = A[M,K] * W[Nc,K]^T    (bf16 in, fp32 accumulate)
// One wave computes a 64x64 tile: 4x4 fragments, 16 WMMAs per K=32 step.
// STORE_BF16 selects the epilogue at compile time (branch-free stores).
// ---------------------------------------------------------------------------
template <bool STORE_BF16>
__global__ __launch_bounds__(256)
void gemm_bf16_wmma(const unsigned short* __restrict__ A,
                    const unsigned short* __restrict__ W,
                    float* __restrict__ Cf,
                    unsigned short* __restrict__ Cb,
                    const float* __restrict__ bias,
                    int M, int Nc, int K) {
    const int lane = threadIdx.x & 31;
    const int wid  = blockIdx.x * (blockDim.x >> 5) + (threadIdx.x >> 5);
    const int ntn  = Nc >> 6;
    const int tm = wid / ntn, tn = wid % ntn;
    if (tm >= (M >> 6)) return;            // uniform per wave
    const int m0 = tm << 6, n0 = tn << 6;

    const unsigned short* Arow[4];
    const unsigned short* Wrow[4];
#pragma unroll
    for (int i = 0; i < 4; ++i) {
        Arow[i] = A + (size_t)(m0 + 16 * i) * K;
        Wrow[i] = W + (size_t)(n0 + 16 * i) * K;
    }

    v8f zero = {};
    v8f acc[4][4];
#pragma unroll
    for (int i = 0; i < 4; ++i)
#pragma unroll
        for (int j = 0; j < 4; ++j) acc[i][j] = zero;

    for (int k0 = 0; k0 < K; k0 += 32) {
        // prefetch next K-slab (global_prefetch_b8); wraps at end (harmless)
        __builtin_prefetch(Arow[0] + (size_t)(lane & 15) * K + ((k0 + 32) & (K - 1)));
        __builtin_prefetch(Wrow[0] + (size_t)(lane & 15) * K + ((k0 + 32) & (K - 1)));

        v16bf a[4], b[4];
#pragma unroll
        for (int i = 0; i < 4; ++i) a[i] = load_a_frag(Arow[i], K, k0, lane);
#pragma unroll
        for (int j = 0; j < 4; ++j) b[j] = load_b_frag(Wrow[j], K, k0, lane);
#pragma unroll
        for (int i = 0; i < 4; ++i)
#pragma unroll
            for (int j = 0; j < 4; ++j)
                acc[i][j] = __builtin_amdgcn_wmma_f32_16x16x32_bf16(
                    false, a[i], false, b[j], (short)0, acc[i][j], false, false);
    }

    // C layout: elem e of lane -> row = e + 8*(lane>>4), col = lane&15
    const int col = lane & 15, hi = lane >> 4;
#pragma unroll
    for (int i = 0; i < 4; ++i)
#pragma unroll
        for (int j = 0; j < 4; ++j) {
            const size_t base = (size_t)(m0 + 16 * i + 8 * hi) * Nc + (n0 + 16 * j + col);
            if (STORE_BF16) {
                unsigned short* p = Cb + base;
#pragma unroll
                for (int e = 0; e < 8; ++e) p[(size_t)e * Nc] = f32_to_bf16(acc[i][j][e]);
            } else {
                const float bj = bias[n0 + 16 * j + col];
                float* p = Cf + base;
#pragma unroll
                for (int e = 0; e < 8; ++e) p[(size_t)e * Nc] = acc[i][j][e] + bj;
            }
        }
}

// ---------------------------------------------------------------------------
// Per-token head attention: one wave per token.
// 1) async-copy v tile (contiguous 2KB) global->LDS (overlaps with 2))
// 2) scores = q(16x64) * k^T : 2 WMMAs; softmax in C-layout registers
// 3) w staged via LDS into an A-fragment
// 4) attn = w(16x16, K zero-padded to 32) * v : 4 WMMAs, v read from LDS
// ---------------------------------------------------------------------------
__global__ __launch_bounds__(256)
void attn_heads_wmma(const unsigned short* __restrict__ qkv,   // [M_TOK, 3D] bf16
                     unsigned short* __restrict__ attn) {      // [M_TOK, D]  bf16
    __shared__ __align__(16) unsigned short vstage[8 * 1024];  // 8 waves * 16x64 bf16
    __shared__ __align__(16) unsigned short wstage[8 * 256];   // 8 waves * 16x16 bf16
    const int lane  = threadIdx.x & 31;
    const int wslot = threadIdx.x >> 5;
    const int tok   = blockIdx.x * 8 + wslot;

    const unsigned short* q = qkv + (size_t)tok * THREE_D;
    const unsigned short* k = q + DD;
    const unsigned short* v = q + 2 * DD;

    // ---- async global->LDS copy of the v tile (16x64 bf16, contiguous 2KB).
    // Each lane moves 4 x 16B; tracked by ASYNCcnt, overlaps scores/softmax.
    unsigned short* vl = vstage + wslot * 1024;
    {
        unsigned       ldsoff = (unsigned)(uintptr_t)vl + (unsigned)lane * 16u;
        unsigned long long ga = (unsigned long long)(uintptr_t)v + (unsigned long long)lane * 16u;
#pragma unroll
        for (int i = 0; i < 4; ++i) {
            asm volatile("global_load_async_to_lds_b128 %0, %1, off"
                         :: "v"(ldsoff + (unsigned)i * 512u),
                            "v"(ga + (unsigned long long)i * 512u)
                         : "memory");
        }
    }

    // ---- scores[h,g] = sum_d q[h,d] k[g,d]  (K = 64 -> 2 WMMA steps)
    v8f s = {};
#pragma unroll
    for (int t = 0; t < 2; ++t) {
        v16bf aq = load_a_frag(q, HDD, t * 32, lane);
        v16bf bk = load_b_frag(k, HDD, t * 32, lane);
        s = __builtin_amdgcn_wmma_f32_16x16x32_bf16(
            false, aq, false, bk, (short)0, s, false, false);
    }

    // ---- softmax over g (columns): row fixed per element, cols across 16
    // lanes. shfl_xor masks 1,2,4,8 stay inside each 16-lane half.
    const int col = lane & 15, hi = lane >> 4;
    unsigned short* wl = wstage + wslot * 256;
#pragma unroll
    for (int e = 0; e < 8; ++e) {
        float x = s[e] * 0.03125f;              // 1/sqrt(1024)
        float m = x;
#pragma unroll
        for (int d = 1; d < 16; d <<= 1) m = fmaxf(m, __shfl_xor(m, d, 32));
        float ex = __expf(x - m);
        float sum = ex;
#pragma unroll
        for (int d = 1; d < 16; d <<= 1) sum += __shfl_xor(sum, d, 32);
        wl[(e + 8 * hi) * 16 + col] = f32_to_bf16(ex / sum);
    }
    // same-wave LDS RAW on wstage
    asm volatile("s_wait_dscnt 0x0" ::: "memory");

    // ---- w as A-fragment, K padded 16 -> 32 with zeros
    v16bf wf;
    {
        v8bf wrow = *(const v8bf*)(wl + (lane & 15) * 16 + hi * 8);
#pragma unroll
        for (int i = 0; i < 8; ++i) { wf[i] = wrow[i]; wf[i + 8] = (__bf16)0.0f; }
    }

    // ---- wait for the async v copy before reading it from LDS
    asm volatile("s_wait_asynccnt 0x0" ::: "memory");

    // attn[h, d0+c] = sum_g w[h,g] v[g, d0+c]; B-frag elems 0..7 = K 8*hi..+8,
    // elems 8..15 = K 16..31 (zero pad). No divergence: load then select.
    unsigned short* dst = attn + (size_t)tok * DD;
#pragma unroll
    for (int j = 0; j < 4; ++j) {
        const int d0 = j * 16;
        v16bf vf;
#pragma unroll
        for (int e = 0; e < 16; ++e) {
            __bf16 val = ((const __bf16*)vl)[((e & 7) + 8 * hi) * HDD + d0 + col];
            vf[e] = (e >= 8) ? (__bf16)0.0f : val;
        }
        v8f o = {};
        o = __builtin_amdgcn_wmma_f32_16x16x32_bf16(
            false, wf, false, vf, (short)0, o, false, false);
#pragma unroll
        for (int e = 0; e < 8; ++e)
            dst[(size_t)(e + 8 * hi) * HDD + d0 + col] = f32_to_bf16(o[e]);
    }
}

// ---------------------------------------------------------------------------
// Launch: convert -> GEMM(qkv) -> head-attention -> GEMM(out)+bias
// Workspace layout (bytes):
//   [0,          32M)  x_bf16    [M_TOK, D]  -- reused as attn_bf16 after GEMM1
//   [32M,        38M)  Wqkv_bf16 [3D, D]
//   [38M,        40M)  Wo_bf16   [D, D]
//   [40M,       136M)  qkv_bf16  [M_TOK, 3D]
// ---------------------------------------------------------------------------
extern "C" void kernel_launch(void* const* d_in, const int* in_sizes, int n_in,
                              void* d_out, int out_size, void* d_ws, size_t ws_size,
                              hipStream_t stream) {
    const float* x    = (const float*)d_in[0];
    const float* Wqkv = (const float*)d_in[1];
    const float* Wo   = (const float*)d_in[2];
    const float* bo   = (const float*)d_in[3];
    float* out = (float*)d_out;

    char* ws = (char*)d_ws;
    unsigned short* xb    = (unsigned short*)(ws);
    unsigned short* wqkvb = (unsigned short*)(ws + (size_t)33554432);
    unsigned short* wob   = (unsigned short*)(ws + (size_t)39845888);
    unsigned short* qkvb  = (unsigned short*)(ws + (size_t)41943040);
    unsigned short* attnb = xb;   // x dead after GEMM1; reuse for attn output

    // 1) fp32 -> bf16 conversions
    {
        int n4 = M_TOK * DD / 4;
        f32_to_bf16_kernel<<<(n4 + 255) / 256, 256, 0, stream>>>(x, xb, n4);
    }
    {
        int n4 = THREE_D * DD / 4;
        f32_to_bf16_kernel<<<(n4 + 255) / 256, 256, 0, stream>>>(Wqkv, wqkvb, n4);
    }
    {
        int n4 = DD * DD / 4;
        f32_to_bf16_kernel<<<(n4 + 255) / 256, 256, 0, stream>>>(Wo, wob, n4);
    }

    // 2) qkv = x * Wqkv^T   (M=16384, N=3072, K=1024), bf16 output
    {
        int waves = (M_TOK / 64) * (THREE_D / 64);          // 12288
        gemm_bf16_wmma<true><<<waves / 8, 256, 0, stream>>>(
            xb, wqkvb, nullptr, qkvb, nullptr, M_TOK, THREE_D, DD);
    }

    // 3) per-token 16x16 head attention
    attn_heads_wmma<<<M_TOK / 8, 256, 0, stream>>>(qkvb, attnb);

    // 4) out = attn * Wo^T + bo   (M=16384, N=1024, K=1024), fp32 output
    {
        int waves = (M_TOK / 64) * (DD / 64);               // 4096
        gemm_bf16_wmma<false><<<waves / 8, 256, 0, stream>>>(
            attnb, wob, out, nullptr, bo, M_TOK, DD, DD);
    }
}
